// CausalSelfAttention_13151189861044
// MI455X (gfx1250) — matrix operand order repaired and checked by reading, not compiled, not run on hardware
//
#include <hip/hip_runtime.h>

#define DEVI __device__ __forceinline__

typedef unsigned short u16;
typedef unsigned int   u32;
typedef unsigned long long u64;
typedef __attribute__((ext_vector_type(16))) __bf16 bf16x16;
typedef __attribute__((ext_vector_type(8)))  __bf16 bf16x8;
typedef __attribute__((ext_vector_type(8)))  float  f32x8;
typedef __attribute__((ext_vector_type(4)))  unsigned int u32x4;
typedef __attribute__((ext_vector_type(8)))  int i32x8;
typedef __attribute__((ext_vector_type(4)))  int i32x4;
typedef __attribute__((ext_vector_type(4)))  u16 u16x4;

// Problem constants (match reference)
#define BB   2
#define TT   2048
#define HH   16
#define DHD  64
#define CC   1024
#define FF3  3072

#if __has_builtin(__builtin_amdgcn_tensor_load_to_lds) && __has_builtin(__builtin_amdgcn_s_wait_tensorcnt)
#define USE_TDM 1
#else
#define USE_TDM 0
#endif

DEVI u16 f32_to_bf16_bits(float f) {
    u32 u = __builtin_bit_cast(u32, f);
    u32 r = u + 0x7FFFu + ((u >> 16) & 1u);   // round-to-nearest-even
    return (u16)(r >> 16);
}

// A-matrix fragment (16x32 bf16): per-lane, 8 elems at p, 8 elems at p+16 (elements)
DEVI bf16x16 load_frag_a(const u16* p) {
    bf16x8 lo = *(const bf16x8*)(p);
    bf16x8 hi = *(const bf16x8*)(p + 16);
    return __builtin_shufflevector(lo, hi, 0,1,2,3,4,5,6,7,8,9,10,11,12,13,14,15);
}
// B-matrix fragment (32x16 bf16): per-lane, 8 elems at p, 8 elems at p+16 (elements)
DEVI bf16x16 load_frag_b(const u16* p) {
    bf16x8 lo = *(const bf16x8*)(p);
    bf16x8 hi = *(const bf16x8*)(p + 16);
    return __builtin_shufflevector(lo, hi, 0,1,2,3,4,5,6,7,8,9,10,11,12,13,14,15);
}

DEVI f32x8 wmma_bf16(bf16x16 a, bf16x16 b, f32x8 c) {
    return __builtin_amdgcn_wmma_f32_16x16x32_bf16(false, a, false, b, (short)0, c, false, false);
}

#if USE_TDM
// Issue a TDM 2-D tile load (tile_rows x tile_cols bf16) from row-major global
// (leading dim ld_elems) into LDS at lds_byte_addr. D# layout per ISA 08 §8.
// This toolchain exposes the 6-arg builtin:
//   (uint32x4 g0, int32x8 g1, int32x4 g2, int32x4 g3, int32x8 g4, i32 cpol)
DEVI void tdm_load_tile_2d(u32 lds_byte_addr, const u16* gsrc,
                           int tile_rows, int tile_cols,
                           int ld_elems, int tensor_rows) {
    u64 ga = (u64)(size_t)(const void*)gsrc;
    u32x4 g0;
    g0[0] = 1u;                                               // count=1, user mode
    g0[1] = lds_byte_addr;                                    // lds_addr
    g0[2] = (u32)ga;                                          // global_addr[31:0]
    g0[3] = ((u32)(ga >> 32) & 0x01FFFFFFu) | (2u << 30);     // addr[56:32], type=2
    u32 td0 = (u32)ld_elems;                                  // tensor_dim0 (elems)
    u32 td1 = (u32)tensor_rows;                               // tensor_dim1
    u64 s0  = (u64)(u32)ld_elems;                             // tensor_dim0_stride
    i32x8 g1;
    g1[0] = (int)(1u << 16);                                  // data_size=1 -> 2 bytes
    g1[1] = (int)((td0 & 0xFFFFu) << 16);                     // atomic_addr=0 | td0.lo16
    g1[2] = (int)(((td0 >> 16) & 0xFFFFu) | ((td1 & 0xFFFFu) << 16));
    g1[3] = (int)(((td1 >> 16) & 0xFFFFu) | (((u32)tile_cols & 0xFFFFu) << 16)); // tile_dim0
    g1[4] = (int)((u32)tile_rows & 0xFFFFu);                  // tile_dim1 | tile_dim2=0
    g1[5] = (int)(u32)(s0 & 0xFFFFFFFFu);                     // stride0[31:0]
    g1[6] = (int)(u32)((s0 >> 32) & 0xFFFFu);                 // stride0[47:32] | stride1.lo=0
    g1[7] = 0;
    i32x4 z4 = {0, 0, 0, 0};
    i32x8 z8 = {0, 0, 0, 0, 0, 0, 0, 0};
    __builtin_amdgcn_tensor_load_to_lds(g0, g1, z4, z4, z8, 0);
}
#endif

// ---------------------------------------------------------------------------
// fp32 -> bf16 conversion (4-wide)
// ---------------------------------------------------------------------------
__global__ void k_cvt(const float* __restrict__ s, u16* __restrict__ d, int n) {
    int i = (blockIdx.x * blockDim.x + threadIdx.x) * 4;
    if (i < n) {   // n is always a multiple of 4 here
        float4 v = *(const float4*)(s + i);
        u16x4 o;
        o.x = f32_to_bf16_bits(v.x);
        o.y = f32_to_bf16_bits(v.y);
        o.z = f32_to_bf16_bits(v.z);
        o.w = f32_to_bf16_bits(v.w);
        *(u16x4*)(d + i) = o;
    }
}

// ---------------------------------------------------------------------------
// LDS-tiled GEMM with TDM panel staging (double buffered):
//   C[M,N] (f32) = A[M,K] (bf16, row-major) * Bw[N,K]^T (bf16, row-major)
// 8 waves / block; macro tile 128(M) x 64(N); wave tile 32x32; K-step 32.
// Wave 0 drives the Tensor Data Mover; everyone computes from LDS.
// ---------------------------------------------------------------------------
__global__ void __launch_bounds__(256) k_gemm_tdm(
        const u16* __restrict__ A, const u16* __restrict__ Bw,
        float* __restrict__ Cm, int M, int N, int K,
        int lda, int ldb, int ldc) {
    __shared__ u16 lA[2][128 * 32];   // A panel: [row][k]  (8 KB each)
    __shared__ u16 lB[2][64 * 32];    // B panel: [n][k]    (4 KB each)

    const int lane = threadIdx.x & 31;
    const int wave = threadIdx.x >> 5;
    const int l16  = lane & 15, hi = lane >> 4;
    const int tilesN = N >> 6;
    const int tm = (blockIdx.x / tilesN) * 128;
    const int tn = (blockIdx.x % tilesN) * 64;
    const int wm = wave >> 1, wn = wave & 1;   // 4 x 2 wave grid

    const u16* aBase = A  + (size_t)tm * lda;
    const u16* bBase = Bw + (size_t)tn * ldb;

    f32x8 acc[2][2] = {};
    int buf = 0;

#if USE_TDM
    if (wave == 0) {
        tdm_load_tile_2d((u32)(size_t)(void*)&lA[0][0], aBase, 128, 32, lda, M);
        tdm_load_tile_2d((u32)(size_t)(void*)&lB[0][0], bBase, 64, 32, ldb, N);
    }
#endif

    for (int k = 0; k < K; k += 32) {
#if USE_TDM
        if (wave == 0) __builtin_amdgcn_s_wait_tensorcnt(0);
        __syncthreads();                 // panels[buf] ready; prev reads done
        if (wave == 0 && (k + 32) < K) {
            tdm_load_tile_2d((u32)(size_t)(void*)&lA[buf ^ 1][0],
                             aBase + (k + 32), 128, 32, lda, M);
            tdm_load_tile_2d((u32)(size_t)(void*)&lB[buf ^ 1][0],
                             bBase + (k + 32), 64, 32, ldb, N);
        }
#else
        __syncthreads();
        // cooperative staging fallback (single buffer)
        for (int c = threadIdx.x; c < 128 * 4; c += 256) {
            int row = c >> 2, col = (c & 3) * 8;
            *(bf16x8*)(&lA[0][row * 32 + col]) =
                *(const bf16x8*)(aBase + (size_t)row * lda + k + col);
        }
        for (int c = threadIdx.x; c < 64 * 4; c += 256) {
            int row = c >> 2, col = (c & 3) * 8;
            *(bf16x8*)(&lB[0][row * 32 + col]) =
                *(const bf16x8*)(bBase + (size_t)row * ldb + k + col);
        }
        __syncthreads();
#endif
        bf16x16 a0 = load_frag_a(&lA[buf][(wm * 32      + l16) * 32 + hi * 8]);
        bf16x16 a1 = load_frag_a(&lA[buf][(wm * 32 + 16 + l16) * 32 + hi * 8]);
        bf16x16 b0 = load_frag_b(&lB[buf][(wn * 32      + l16) * 32 + hi * 8]);
        bf16x16 b1 = load_frag_b(&lB[buf][(wn * 32 + 16 + l16) * 32 + hi * 8]);
        acc[0][0] = wmma_bf16(a0, b0, acc[0][0]);
        acc[0][1] = wmma_bf16(a0, b1, acc[0][1]);
        acc[1][0] = wmma_bf16(a1, b0, acc[1][0]);
        acc[1][1] = wmma_bf16(a1, b1, acc[1][1]);
#if USE_TDM
        buf ^= 1;
#endif
    }

#pragma unroll
    for (int mi = 0; mi < 2; ++mi)
#pragma unroll
        for (int ni = 0; ni < 2; ++ni) {
            float* out = Cm + (size_t)(tm + wm * 32 + mi * 16 + hi * 8) * ldc
                            + tn + wn * 32 + ni * 16 + l16;
#pragma unroll
            for (int r = 0; r < 8; ++r) out[(size_t)r * ldc] = acc[mi][ni][r];
        }
}

// ---------------------------------------------------------------------------
// Per-(b,h,t) row of 64: QK RMSNorm + RoPE; emit bf16 Q,K [bh,T,64] and
// V transposed [bh,64,T]. One wave per row; lane owns d=lane and d=lane+32
// (exactly the RoPE rotation pair).
// ---------------------------------------------------------------------------
__global__ void k_norm_rope(const float* __restrict__ QKV, u16* __restrict__ Q,
                            u16* __restrict__ Kk, u16* __restrict__ Vt) {
    const int lane = threadIdx.x & 31;
    const int wave = threadIdx.x >> 5;
    const int rid  = blockIdx.x * (blockDim.x >> 5) + wave;   // 0 .. B*H*T-1
    const int t  = rid % TT;
    const int bh = rid / TT;
    const int b_ = bh / HH, h_ = bh % HH;

    const float* base = QKV + (size_t)(b_ * TT + t) * FF3 + h_ * DHD;
    float q1 = base[lane],          q2 = base[lane + 32];
    float k1 = base[CC + lane],     k2 = base[CC + lane + 32];
    float v1 = base[2 * CC + lane], v2 = base[2 * CC + lane + 32];

    float qs = q1 * q1 + q2 * q2;
    float ks = k1 * k1 + k2 * k2;
#pragma unroll
    for (int mk = 1; mk < 32; mk <<= 1) {
        qs += __shfl_xor(qs, mk);
        ks += __shfl_xor(ks, mk);
    }
    const float eps = 1.1920929e-7f;
    float qr = rsqrtf(qs * (1.0f / 64.0f) + eps);
    float kr = rsqrtf(ks * (1.0f / 64.0f) + eps);
    q1 *= qr; q2 *= qr; k1 *= kr; k2 *= kr;

    // inv_freq[j] = 10000^(-j/32), angle = t * inv_freq
    float invf = __powf(10000.0f, -(float)lane * (1.0f / 32.0f));
    float ang  = (float)t * invf;
    float cs = cosf(ang), sn = sinf(ang);
    float q1r = q1 * cs - q2 * sn, q2r = q2 * cs + q1 * sn;
    float k1r = k1 * cs - k2 * sn, k2r = k2 * cs + k1 * sn;

    size_t ro = ((size_t)bh * TT + t) * DHD;
    Q [ro + lane]      = f32_to_bf16_bits(q1r);
    Q [ro + lane + 32] = f32_to_bf16_bits(q2r);
    Kk[ro + lane]      = f32_to_bf16_bits(k1r);
    Kk[ro + lane + 32] = f32_to_bf16_bits(k2r);
    Vt[((size_t)bh * DHD + lane)      * TT + t] = f32_to_bf16_bits(v1);
    Vt[((size_t)bh * DHD + lane + 32) * TT + t] = f32_to_bf16_bits(v2);
}

// ---------------------------------------------------------------------------
// Flash attention: one wave per 16-row q-tile; 32-key blocks.
// scores via 4 WMMAs, online softmax in f32 (C-layout row stats, shfl_xor
// within 16-lane halves), P transposed C->A layout through LDS, then 4 PV
// WMMAs against V^T. Output written bf16 [B*T, 1024] for the projection GEMM.
// ---------------------------------------------------------------------------
__global__ void k_attn(const u16* __restrict__ Q, const u16* __restrict__ Kk,
                       const u16* __restrict__ Vt, u16* __restrict__ Y) {
    __shared__ u16 ldsP[4][16 * 32];
    const int lane = threadIdx.x & 31;
    const int wave = threadIdx.x >> 5;
    const int tile = blockIdx.x * 4 + wave;
    const int qTiles = TT / 16;
    const int bh = tile / qTiles;
    const int qt = tile % qTiles;
    const int qBase = qt * 16;
    const int l16 = lane & 15, hi = lane >> 4;
    const int rowLoc = hi * 8;
    u16* P = ldsP[wave];

    const u16* qp = Q + ((size_t)bh * TT + qBase + l16) * DHD + hi * 8;
    bf16x16 qa0 = load_frag_a(qp);
    bf16x16 qa1 = load_frag_a(qp + 32);

    float m[8], sacc[8];
    f32x8 acc[4] = {};
#pragma unroll
    for (int r = 0; r < 8; ++r) { m[r] = -__builtin_inff(); sacc[r] = 0.0f; }

    const float scale = 0.125f;   // 1/sqrt(64)

    for (int kb = 0; kb <= qBase + 15; kb += 32) {
        const u16* kp0 = Kk + ((size_t)bh * TT + kb + l16) * DHD + hi * 8;
        const u16* kp1 = Kk + ((size_t)bh * TT + kb + 16 + l16) * DHD + hi * 8;
        f32x8 s0 = {}, s1 = {};
        s0 = wmma_bf16(qa0, load_frag_b(kp0),      s0);
        s0 = wmma_bf16(qa1, load_frag_b(kp0 + 32), s0);
        s1 = wmma_bf16(qa0, load_frag_b(kp1),      s1);
        s1 = wmma_bf16(qa1, load_frag_b(kp1 + 32), s1);

        float rmax[8];
#pragma unroll
        for (int r = 0; r < 8; ++r) {
            int row = qBase + rowLoc + r;
            float a0 = s0[r] * scale;
            float a1 = s1[r] * scale;
            if (kb + l16 > row)      a0 = -__builtin_inff();
            if (kb + 16 + l16 > row) a1 = -__builtin_inff();
            s0[r] = a0; s1[r] = a1;
            float v = fmaxf(a0, a1);
            v = fmaxf(v, __shfl_xor(v, 1));
            v = fmaxf(v, __shfl_xor(v, 2));
            v = fmaxf(v, __shfl_xor(v, 4));
            v = fmaxf(v, __shfl_xor(v, 8));
            rmax[r] = v;
        }

#pragma unroll
        for (int r = 0; r < 8; ++r) {
            float mn    = fmaxf(m[r], rmax[r]);
            float alpha = __expf(m[r] - mn);
            float p0 = __expf(s0[r] - mn);
            float p1 = __expf(s1[r] - mn);
            float rs = p0 + p1;
            rs += __shfl_xor(rs, 1);
            rs += __shfl_xor(rs, 2);
            rs += __shfl_xor(rs, 4);
            rs += __shfl_xor(rs, 8);
            sacc[r] = sacc[r] * alpha + rs;
            m[r] = mn;
            acc[0][r] *= alpha; acc[1][r] *= alpha;
            acc[2][r] *= alpha; acc[3][r] *= alpha;
            P[(rowLoc + r) * 32 + l16]      = f32_to_bf16_bits(p0);
            P[(rowLoc + r) * 32 + l16 + 16] = f32_to_bf16_bits(p1);
        }
        asm volatile("s_wait_dscnt 0" ::: "memory");

        bf16x16 pa = load_frag_a(P + l16 * 32 + hi * 8);

#pragma unroll
        for (int dt = 0; dt < 4; ++dt) {
            const u16* vp = Vt + ((size_t)bh * DHD + dt * 16 + l16) * TT + kb + hi * 8;
            acc[dt] = wmma_bf16(pa, load_frag_b(vp), acc[dt]);
        }
    }

    const int b_ = bh / HH, h_ = bh % HH;
#pragma unroll
    for (int dt = 0; dt < 4; ++dt) {
        int d = dt * 16 + l16;
#pragma unroll
        for (int r = 0; r < 8; ++r) {
            int row = qBase + rowLoc + r;
            float y = acc[dt][r] / sacc[r];
            Y[((size_t)(b_ * TT + row)) * CC + h_ * DHD + d] = f32_to_bf16_bits(y);
        }
    }
}

// ---------------------------------------------------------------------------
// Host-side orchestration
// ---------------------------------------------------------------------------
extern "C" void kernel_launch(void* const* d_in, const int* in_sizes, int n_in,
                              void* d_out, int out_size, void* d_ws, size_t ws_size,
                              hipStream_t stream) {
    (void)in_sizes; (void)n_in; (void)out_size; (void)ws_size;
    const float* x      = (const float*)d_in[0];
    const float* w_attn = (const float*)d_in[1];
    const float* w_proj = (const float*)d_in[2];
    float* out = (float*)d_out;

    char* ws = (char*)d_ws;
    size_t off = 0;
    u16*   x_bf  = (u16*)(ws + off); off += (size_t)BB * TT * CC * 2;   //  8 MB
    u16*   wA_bf = (u16*)(ws + off); off += (size_t)FF3 * CC * 2;       //  6 MB
    u16*   wP_bf = (u16*)(ws + off); off += (size_t)CC * CC * 2;        //  2 MB
    float* qkv   = (float*)(ws + off); off += (size_t)BB * TT * FF3 * 4;// 48 MB
    u16*   q_bf  = (u16*)(ws + off); off += (size_t)BB * HH * TT * DHD * 2; // 8 MB
    u16*   k_bf  = (u16*)(ws + off); off += (size_t)BB * HH * TT * DHD * 2; // 8 MB
    u16*   vT_bf = (u16*)(ws + off); off += (size_t)BB * HH * TT * DHD * 2; // 8 MB
    u16*   y_bf  = (u16*)(ws + off); off += (size_t)BB * TT * CC * 2;   //  8 MB

    const int nX = BB * TT * CC, nWA = FF3 * CC, nWP = CC * CC;
    k_cvt<<<(nX  / 4 + 255) / 256, 256, 0, stream>>>(x,      x_bf,  nX);
    k_cvt<<<(nWA / 4 + 255) / 256, 256, 0, stream>>>(w_attn, wA_bf, nWA);
    k_cvt<<<(nWP / 4 + 255) / 256, 256, 0, stream>>>(w_proj, wP_bf, nWP);

    // QKV GEMM: [4096,3072] = [4096,1024] x [3072,1024]^T
    k_gemm_tdm<<<(BB * TT / 128) * (FF3 / 64), 256, 0, stream>>>(
        x_bf, wA_bf, qkv, BB * TT, FF3, CC, CC, CC, FF3);

    // RMSNorm + RoPE + head-split/transpose
    k_norm_rope<<<(BB * HH * TT) / 8, 256, 0, stream>>>(qkv, q_bf, k_bf, vT_bf);

    // Flash attention
    k_attn<<<(BB * HH * (TT / 16)) / 4, 128, 0, stream>>>(q_bf, k_bf, vT_bf, y_bf);

    // Output projection: [4096,1024] = [4096,1024] x [1024,1024]^T -> fp32 out
    k_gemm_tdm<<<(BB * TT / 128) * (CC / 64), 256, 0, stream>>>(
        y_bf, wP_bf, out, BB * TT, CC, CC, CC, CC, CC);
}
